// FlexibleGNN_18674517803529
// MI455X (gfx1250) — matrix operand order repaired and checked by reading, not compiled, not run on hardware
//
#include <hip/hip_runtime.h>
#include <hip/hip_bf16.h>
#include <math.h>

// ---------------------------------------------------------------------------
// Types
// ---------------------------------------------------------------------------
typedef __attribute__((ext_vector_type(16))) _Float16 v16h;
typedef __attribute__((ext_vector_type(8)))  float    v8f;
typedef __attribute__((ext_vector_type(4)))  float    f32x4;

#define D_DIM 128
#define NEG_SLOPE 0.2f

// ---------------------------------------------------------------------------
// Helpers
// ---------------------------------------------------------------------------
__device__ __forceinline__ unsigned enc_f(float f) {
    unsigned u = __float_as_uint(f);
    return (u & 0x80000000u) ? ~u : (u | 0x80000000u);
}
__device__ __forceinline__ float dec_f(unsigned u) {
    return (u & 0x80000000u) ? __uint_as_float(u & 0x7FFFFFFFu)
                             : __uint_as_float(~u);
}

// A fragment: 16x32 f16, per ISA 7.12.2 (16-bit A layout).
// lane L: row M = L%16; halves[0..7]  = K kstep + (L>=16?8:0) .. +7
//                       halves[8..15] = K kstep+16+(L>=16?8:0) .. +7
__device__ __forceinline__ v16h load_a(const _Float16* __restrict__ rowp,
                                       int ks, int hi) {
    union { v16h v; f32x4 q[2]; } u;
    u.q[0] = *(const f32x4*)(rowp + ks + hi * 8);
    u.q[1] = *(const f32x4*)(rowp + ks + 16 + hi * 8);
    return u.v;
}

// B fragment: 32x16 f16. lane L: col N = L%16, halves[0..7] = K kstep+(L>=16?8:0)..+7,
// halves[8..15] = K kstep+16+(L>=16?8:0)..+7.  B[k][n] = W[n][k] -> contiguous
// slice of W row n (row-major), so no transpose needed.
__device__ __forceinline__ v16h load_b(const _Float16* __restrict__ rowp,
                                       int ks, int hi) {
    union { v16h v; f32x4 q[2]; } u;
    const _Float16* p = rowp + ks + hi * 8;
    u.q[0] = *(const f32x4*)(p);
    u.q[1] = *(const f32x4*)(p + 16);
    return u.v;
}

// ---------------------------------------------------------------------------
// fp32 -> (f16 hi, f16 lo) split (used for x, h, and all W matrices)
// ---------------------------------------------------------------------------
__global__ void split_f32_f16(const float* __restrict__ src,
                              _Float16* __restrict__ hi,
                              _Float16* __restrict__ lo, int n) {
    int i = blockIdx.x * blockDim.x + threadIdx.x;
    if (i >= n) return;
    float v = src[i];
    _Float16 h = (_Float16)v;
    hi[i] = h;
    lo[i] = (_Float16)(v - (float)h);
}

// ---------------------------------------------------------------------------
// WMMA GEMM:  out[m,n] = sum_k h[m,k] * W[n,k]   (i.e. h @ W^T)
// Split-f16 precision: C = Ah*Bh + Ah*Bl + Al*Bh  (f32 accumulate)
// grid = (N/16, 2), block = 256 (8 waves; wave w owns N-tile w of 8)
// ---------------------------------------------------------------------------
__global__ void gemm_split_wmma(const _Float16* __restrict__ xh,
                                const _Float16* __restrict__ xlo,
                                const _Float16* __restrict__ WhL,
                                const _Float16* __restrict__ WloL,
                                const _Float16* __restrict__ WhR,
                                const _Float16* __restrict__ WloR,
                                float* __restrict__ xl,
                                float* __restrict__ xr) {
    const int mtile = blockIdx.x;
    const int mat   = blockIdx.y;
    const _Float16* __restrict__ Wh  = mat ? WhR  : WhL;
    const _Float16* __restrict__ Wlo = mat ? WloR : WloL;
    float* __restrict__ out = mat ? xr : xl;

    const int wave = threadIdx.x >> 5;      // n-tile (0..7)
    const int lane = threadIdx.x & 31;
    const int row  = lane & 15;
    const int hi   = lane >> 4;

    const _Float16* arow_h  = xh  + (size_t)(mtile * 16 + row) * D_DIM;
    const _Float16* arow_lo = xlo + (size_t)(mtile * 16 + row) * D_DIM;
    const int ncol = wave * 16 + row;
    const _Float16* brow_h  = Wh  + (size_t)ncol * D_DIM;
    const _Float16* brow_lo = Wlo + (size_t)ncol * D_DIM;

    v8f c = {};
#pragma unroll
    for (int ks = 0; ks < D_DIM; ks += 32) {
        v16h ah = load_a(arow_h,  ks, hi);
        v16h al = load_a(arow_lo, ks, hi);
        v16h bh = load_b(brow_h,  ks, hi);
        v16h bl = load_b(brow_lo, ks, hi);
        c = __builtin_amdgcn_wmma_f32_16x16x32_f16(false, ah, false, bh,
                                                   (short)0, c, false, false);
        c = __builtin_amdgcn_wmma_f32_16x16x32_f16(false, ah, false, bl,
                                                   (short)0, c, false, false);
        c = __builtin_amdgcn_wmma_f32_16x16x32_f16(false, al, false, bh,
                                                   (short)0, c, false, false);
    }

    // C/D layout: VGPR r, lanes 0-15: (M = r,   N = lane)
    //                      lanes 16-31:(M = r+8, N = lane-16)
    const int ncol_out = wave * 16 + (lane & 15);
#pragma unroll
    for (int r = 0; r < 8; ++r) {
        int m = mtile * 16 + r + hi * 8;
        out[(size_t)m * D_DIM + ncol_out] = c[r];
    }
}

// ---------------------------------------------------------------------------
// Per-layer init: acc = 0, emax = enc(-inf) ~ 0, denom = 0
// ---------------------------------------------------------------------------
__global__ void init_layer(float* __restrict__ acc,
                           unsigned* __restrict__ emax_u,
                           float* __restrict__ denom, int nelem, int nnode) {
    int i = blockIdx.x * blockDim.x + threadIdx.x;
    if (i < nelem) acc[i] = 0.0f;
    if (i < nnode) { emax_u[i] = 0u; denom[i] = 0.0f; }
}

// ---------------------------------------------------------------------------
// Edge logits: one wave per edge.
// m = leakyrelu(xl[src]+xr[dst]); e = m . att ; atomicMax(emax[dst], e)
// ---------------------------------------------------------------------------
__global__ void edge_logits(const float* __restrict__ xl,
                            const float* __restrict__ xr,
                            const int* __restrict__ ei,
                            const float* __restrict__ att,
                            float* __restrict__ e_buf,
                            unsigned* __restrict__ emax_u, int E, int EP) {
    int gid  = blockIdx.x * blockDim.x + threadIdx.x;
    int eid  = gid >> 5;
    int lane = gid & 31;
    if (eid >= EP) return;
    int src, dst;
    if (eid < E) { src = ei[eid]; dst = ei[E + eid]; }
    else         { src = dst = eid - E; }

    f32x4 a  = *(const f32x4*)(xl  + (size_t)src * D_DIM + lane * 4);
    f32x4 b  = *(const f32x4*)(xr  + (size_t)dst * D_DIM + lane * 4);
    f32x4 av = *(const f32x4*)(att + lane * 4);
    f32x4 t  = a + b;
    float p = 0.0f;
#pragma unroll
    for (int j = 0; j < 4; ++j) {
        float m = t[j];
        m = (m > 0.0f) ? m : NEG_SLOPE * m;
        p += m * av[j];
    }
#pragma unroll
    for (int off = 16; off > 0; off >>= 1) p += __shfl_xor(p, off, 32);
    if (lane == 0) {
        e_buf[eid] = p;
        atomicMax(emax_u + dst, enc_f(p));
    }
}

// ---------------------------------------------------------------------------
// ex = exp(e - emax[dst]);  denom[dst] += ex.   One thread per edge.
// ---------------------------------------------------------------------------
__global__ void edge_expsum(const float* __restrict__ e_buf,
                            const unsigned* __restrict__ emax_u,
                            const int* __restrict__ ei,
                            float* __restrict__ ex_buf,
                            float* __restrict__ denom, int E, int EP) {
    int eid = blockIdx.x * blockDim.x + threadIdx.x;
    if (eid >= EP) return;
    int dst = (eid < E) ? ei[E + eid] : (eid - E);
    float ex = expf(e_buf[eid] - dec_f(emax_u[dst]));
    ex_buf[eid] = ex;
    __hip_atomic_fetch_add(denom + dst, ex, __ATOMIC_RELAXED,
                           __HIP_MEMORY_SCOPE_AGENT);
}

// ---------------------------------------------------------------------------
// acc[dst] += (ex/denom[dst]) * xl[src].   One wave per edge.
// ---------------------------------------------------------------------------
__global__ void edge_scatter(const float* __restrict__ xl,
                             const int* __restrict__ ei,
                             const float* __restrict__ ex_buf,
                             const float* __restrict__ denom,
                             float* __restrict__ acc, int E, int EP) {
    int gid  = blockIdx.x * blockDim.x + threadIdx.x;
    int eid  = gid >> 5;
    int lane = gid & 31;
    if (eid >= EP) return;
    int src, dst;
    if (eid < E) { src = ei[eid]; dst = ei[E + eid]; }
    else         { src = dst = eid - E; }
    float alpha = ex_buf[eid] / denom[dst];
    f32x4 v = *(const f32x4*)(xl + (size_t)src * D_DIM + lane * 4);
    float* ap = acc + (size_t)dst * D_DIM + lane * 4;
#pragma unroll
    for (int j = 0; j < 4; ++j)
        __hip_atomic_fetch_add(ap + j, v[j] * alpha, __ATOMIC_RELAXED,
                               __HIP_MEMORY_SCOPE_AGENT);
}

// ---------------------------------------------------------------------------
// h = gelu(acc + bias); emit f16 hi/lo split for next layer GEMM;
// optionally write fp32 result to d_out (last layer).
// ---------------------------------------------------------------------------
__global__ void finalize_layer(const float* __restrict__ acc,
                               const float* __restrict__ bias,
                               _Float16* __restrict__ xh,
                               _Float16* __restrict__ xlo,
                               float* __restrict__ out, int nelem) {
    int i = blockIdx.x * blockDim.x + threadIdx.x;
    if (i >= nelem) return;
    float t = acc[i] + bias[i & (D_DIM - 1)];
    float g = 0.5f * t * (1.0f + erff(t * 0.70710678118654752f));
    _Float16 h = (_Float16)g;
    xh[i]  = h;
    xlo[i] = (_Float16)(g - (float)h);
    if (out) out[i] = g;
}

// ---------------------------------------------------------------------------
// Host launch
// ---------------------------------------------------------------------------
extern "C" void kernel_launch(void* const* d_in, const int* in_sizes, int n_in,
                              void* d_out, int out_size, void* d_ws,
                              size_t ws_size, hipStream_t stream) {
    (void)n_in; (void)out_size; (void)ws_size;
    const int N  = in_sizes[0] / D_DIM;   // 50000
    const int E  = in_sizes[1] / 2;       // 800000
    const int EP = E + N;                 // with self loops
    const int NE = N * D_DIM;

    const float* x  = (const float*)d_in[0];
    const int*   ei = (const int*)d_in[1];

    // ---- workspace carve (256B aligned) ----
    char* ws = (char*)d_ws;
    auto carve = [&](size_t bytes) {
        char* p = ws;
        ws += (bytes + 255) & ~(size_t)255;
        return p;
    };
    _Float16* xh     = (_Float16*)carve((size_t)NE * 2);
    _Float16* xlo    = (_Float16*)carve((size_t)NE * 2);
    float*    xl     = (float*)   carve((size_t)NE * 4);
    float*    xr     = (float*)   carve((size_t)NE * 4);
    float*    acc    = (float*)   carve((size_t)NE * 4);
    float*    e_buf  = (float*)   carve((size_t)EP * 4);
    float*    ex_buf = (float*)   carve((size_t)EP * 4);
    unsigned* emax_u = (unsigned*)carve((size_t)N * 4);
    float*    denom  = (float*)   carve((size_t)N * 4);
    _Float16* Wh[6];
    _Float16* Wlo[6];
    for (int i = 0; i < 6; ++i) {
        Wh[i]  = (_Float16*)carve((size_t)D_DIM * D_DIM * 2);
        Wlo[i] = (_Float16*)carve((size_t)D_DIM * D_DIM * 2);
    }

    const int WN = D_DIM * D_DIM;
    // ---- split all 6 weight matrices to f16 hi/lo ----
    for (int l = 0; l < 3; ++l) {
        const float* WlP = (const float*)d_in[2 + 4 * l];
        const float* WrP = (const float*)d_in[3 + 4 * l];
        split_f32_f16<<<(WN + 255) / 256, 256, 0, stream>>>(WlP, Wh[2 * l],
                                                            Wlo[2 * l], WN);
        split_f32_f16<<<(WN + 255) / 256, 256, 0, stream>>>(WrP, Wh[2 * l + 1],
                                                            Wlo[2 * l + 1], WN);
    }
    // ---- split input features ----
    split_f32_f16<<<(NE + 255) / 256, 256, 0, stream>>>(x, xh, xlo, NE);

    const int mtiles   = N / 16;
    const int edgeGrid = (EP * 32 + 255) / 256;
    for (int l = 0; l < 3; ++l) {
        const float* att  = (const float*)d_in[4 + 4 * l];
        const float* bias = (const float*)d_in[5 + 4 * l];

        gemm_split_wmma<<<dim3(mtiles, 2), 256, 0, stream>>>(
            xh, xlo, Wh[2 * l], Wlo[2 * l], Wh[2 * l + 1], Wlo[2 * l + 1], xl,
            xr);
        init_layer<<<(NE + 255) / 256, 256, 0, stream>>>(acc, emax_u, denom,
                                                         NE, N);
        edge_logits<<<edgeGrid, 256, 0, stream>>>(xl, xr, ei, att, e_buf,
                                                  emax_u, E, EP);
        edge_expsum<<<(EP + 255) / 256, 256, 0, stream>>>(e_buf, emax_u, ei,
                                                          ex_buf, denom, E, EP);
        edge_scatter<<<edgeGrid, 256, 0, stream>>>(xl, ei, ex_buf, denom, acc,
                                                   E, EP);
        finalize_layer<<<(NE + 255) / 256, 256, 0, stream>>>(
            acc, bias, xh, xlo, (l == 2) ? (float*)d_out : nullptr, NE);
    }
}
